// ResNeXtBottleneckB_85212151152993
// MI455X (gfx1250) — matrix operand order repaired and checked by reading, not compiled, not run on hardware
//
#include <hip/hip_runtime.h>
#include <stddef.h>

// ---------------------------------------------------------------------------
// ResNeXt bottleneck (cardinality 32) fused pipeline for gfx1250 (MI455X).
//   conv1 1x1 256->512  + BN(batch stats) + ReLU
//   conv3 3x3 grouped(32) 512->512 + BN + ReLU
//   convA 1x1 512->256  + BN + residual + ReLU
// Memory-bound (~1.2GB HBM traffic @ 23.3TB/s ~ 52us): bf16 storage for
// weights/activations, v_wmma_f32_16x16x32_bf16 with f32 accumulation.
// All intermediates (51-103MB) are L2-resident (192MB L2), so the multi-pass
// re-reads (conv taps, per-co-tile K re-reads) hit L2, not HBM.
// x is pre-converted once to bf16 [c][P] layout so every GEMM's B-tile load
// is a uniform strided u16 load with no in-loop conversion VALU.
// ---------------------------------------------------------------------------

typedef __attribute__((ext_vector_type(16))) __bf16 bfvec16;
typedef __attribute__((ext_vector_type(8)))  float  fvec8;

#define NPOS   100352   // 32 * 56 * 56
#define IMGHW  3136     // 56 * 56
#define HW_W   56

// Native f32 -> bf16 (RNE): let the backend pick v_cvt_*bf16* forms.
static __device__ __forceinline__ unsigned short f2bfu(float f) {
  return __builtin_bit_cast(unsigned short, static_cast<__bf16>(f));
}
static __device__ __forceinline__ __bf16 bits2bf(unsigned short s) {
  return __builtin_bit_cast(__bf16, s);
}
static __device__ __forceinline__ float bf2f(unsigned short s) {
  return __builtin_bit_cast(float, (unsigned int)s << 16);
}

// ---------------------------------------------------------------------------
// K0a: convert weights to bf16 (+ repack w3 to [cd][tap][ci], pad K 144->160),
//      zero the stat accumulators.
// ---------------------------------------------------------------------------
__global__ __launch_bounds__(256) void prep_kernel(
    const float* __restrict__ w1, const float* __restrict__ w3,
    const float* __restrict__ wa, unsigned short* __restrict__ w1b,
    unsigned short* __restrict__ w3p, unsigned short* __restrict__ wab,
    float* __restrict__ stats) {
  const int i = blockIdx.x * 256 + threadIdx.x;
  if (i < 131072) {              // 512*256 == 256*512
    w1b[i] = f2bfu(w1[i]);
    wab[i] = f2bfu(wa[i]);
  }
  if (i < 81920) {               // 512 * 160
    const int cd = i / 160;
    const int k  = i - cd * 160; // k = tap*16 + ci, padded to 160
    unsigned short v = 0;
    if (k < 144) {
      const int ci  = k & 15;
      const int tap = k >> 4;    // kh*3 + kw
      v = f2bfu(w3[cd * 144 + ci * 9 + tap]);
    }
    w3p[i] = v;
  }
  if (i < 2560) stats[i] = 0.0f;
}

// ---------------------------------------------------------------------------
// K0b: x (NCHW fp32) -> xb (bf16, [c][P] with P = n*3136 + hw). grid (392,256).
// ---------------------------------------------------------------------------
__global__ __launch_bounds__(256) void xcvt_kernel(
    const float* __restrict__ x, unsigned short* __restrict__ xb) {
  const int c = blockIdx.y;
  const int p = blockIdx.x * 256 + threadIdx.x;   // < 100352 exactly
  const int n  = p / IMGHW;
  const int hw = p - n * IMGHW;
  xb[(size_t)c * NPOS + p] = f2bfu(x[((size_t)n * 256 + c) * IMGHW + hw]);
}

// ---------------------------------------------------------------------------
// WMMA operand packing (contiguous k order, same in A and B) for v_wmma_f32_16x16x32_bf16:
//  A (16xK row per lane): elem e -> K = kb + e,
//                         kb = s*32 + (lane>=16 ? 16 : 0)
//  B (Kx16): N = lane&15; elem e -> K = s*32 + (lane>>4)*16 + e
//  D (16x16 f32): N = lane&15; vgpr r -> M = r + (lane>>4)*8
// ---------------------------------------------------------------------------
static __device__ __forceinline__ bfvec16 load_a_tile(
    const unsigned short* __restrict__ wrow, int kb) {
  bfvec16 a;
#pragma unroll
  for (int e = 0; e < 16; ++e) {
    const int k = kb + e;
    a[e] = bits2bf(wrow[k]);
  }
  return a;
}

// B tile from a [C][NPOS] bf16 activation buffer (column = one position).
static __device__ __forceinline__ bfvec16 load_b_tile(
    const unsigned short* __restrict__ act, int krow_base, int pos) {
  const unsigned short* bp = act + (size_t)krow_base * NPOS + pos;
  bfvec16 b;
#pragma unroll
  for (int e = 0; e < 16; ++e) b[e] = bits2bf(bp[(size_t)e * NPOS]);
  return b;
}

static __device__ __forceinline__ void store_and_stat(
    fvec8 acc, unsigned short* __restrict__ y, int co_base, int half, int pos,
    float* sAcc, float* qAcc) {
  unsigned short* yp = y + (size_t)(co_base + (half << 3)) * NPOS + pos;
#pragma unroll
  for (int r = 0; r < 8; ++r) {
    const float v = acc[r];
    yp[(size_t)r * NPOS] = f2bfu(v);
    sAcc[r] += v;
    qAcc[r] += v * v;
  }
}

static __device__ __forceinline__ void flush_stats(
    float* sAcc, float* qAcc, float* __restrict__ sum, float* __restrict__ sq,
    int co_base, int half, int lm) {
#pragma unroll
  for (int r = 0; r < 8; ++r) {
    float v = sAcc[r], q = qAcc[r];
#pragma unroll
    for (int off = 8; off; off >>= 1) {
      v += __shfl_xor(v, off, 16);
      q += __shfl_xor(q, off, 16);
    }
    if (lm == 0) {
      const int c = co_base + (half << 3) + r;
      atomicAdd(&sum[c], v);
      atomicAdd(&sq[c], q);
    }
  }
}

// ---------------------------------------------------------------------------
// K1: conv1 1x1 (M=512, K=256, N=100352). One wave per block; block tile =
//     16 co x 256 pos (16 WMMA N-tiles, 8 K-steps each). A hoisted (64 VGPRs).
// ---------------------------------------------------------------------------
__global__ __launch_bounds__(32) void conv1_kernel(
    const unsigned short* __restrict__ xb, const unsigned short* __restrict__ w1b,
    unsigned short* __restrict__ y1, float* __restrict__ sum1,
    float* __restrict__ sq1) {
  const int lane = threadIdx.x & 31;
  const int half = lane >> 4;
  const int lm   = lane & 15;
  const int co_base = blockIdx.y * 16;
  const int pos0    = blockIdx.x * 256;

  bfvec16 A[8];
  const unsigned short* wrow = w1b + (size_t)(co_base + lm) * 256;
#pragma unroll
  for (int s = 0; s < 8; ++s) A[s] = load_a_tile(wrow, s * 32 + (half << 4));

  float sAcc[8] = {0, 0, 0, 0, 0, 0, 0, 0};
  float qAcc[8] = {0, 0, 0, 0, 0, 0, 0, 0};

  for (int t = 0; t < 16; ++t) {
    const int pos = pos0 + t * 16 + lm;
    // prefetch next tile's B column into cache (global_prefetch_b8)
    __builtin_prefetch(xb + pos + 16, 0, 3);
    fvec8 acc = {};
#pragma unroll
    for (int s = 0; s < 8; ++s) {
      bfvec16 B = load_b_tile(xb, s * 32 + (half << 4), pos);
      acc = __builtin_amdgcn_wmma_f32_16x16x32_bf16(false, A[s], false, B,
                                                    (short)0, acc, false, false);
    }
    store_and_stat(acc, y1, co_base, half, pos, sAcc, qAcc);
  }
  flush_stats(sAcc, qAcc, sum1, sq1, co_base, half, lm);
}

// ---------------------------------------------------------------------------
// K2: BN params. scale = g*rsqrt(var+eps); shift = b - mean*scale.
// ---------------------------------------------------------------------------
__global__ __launch_bounds__(256) void bnparam_kernel(
    const float* __restrict__ sum, const float* __restrict__ sq,
    const float* __restrict__ gamma, const float* __restrict__ beta,
    float* __restrict__ scale, float* __restrict__ shift, int C) {
  const int c = blockIdx.x * 256 + threadIdx.x;
  if (c >= C) return;
  const float inv_n = 1.0f / (float)NPOS;
  const float mean  = sum[c] * inv_n;
  float var = sq[c] * inv_n - mean * mean;
  var = fmaxf(var, 0.0f);
  const float sc = gamma[c] * rsqrtf(var + 1e-5f);
  scale[c] = sc;
  shift[c] = beta[c] - mean * sc;
}

// ---------------------------------------------------------------------------
// K2b: in-place BN + ReLU over a bf16 [C][NPOS] buffer. grid = (392, C).
// ---------------------------------------------------------------------------
__global__ __launch_bounds__(256) void bnapply_kernel(
    unsigned short* __restrict__ y, const float* __restrict__ scale,
    const float* __restrict__ shift) {
  const int c = blockIdx.y;
  const int p = blockIdx.x * 256 + threadIdx.x;
  const size_t idx = (size_t)c * NPOS + p;
  const float v = fmaxf(scale[c] * bf2f(y[idx]) + shift[c], 0.0f);
  y[idx] = f2bfu(v);
}

// ---------------------------------------------------------------------------
// K3: grouped 3x3 conv. Group g == co-tile (16 ch). K = 9 taps * 16 ci = 144,
//     padded to 160 -> 5 WMMA K-steps (tap pairs; A tail is zero).
// ---------------------------------------------------------------------------
__global__ __launch_bounds__(32) void conv3_kernel(
    const unsigned short* __restrict__ a1, const unsigned short* __restrict__ w3p,
    unsigned short* __restrict__ y3, float* __restrict__ sum3,
    float* __restrict__ sq3) {
  const int lane = threadIdx.x & 31;
  const int half = lane >> 4;
  const int lm   = lane & 15;
  const int g       = blockIdx.y;      // group == 16-channel tile
  const int co_base = g * 16;
  const int pos0    = blockIdx.x * 256;

  bfvec16 A[5];
  const unsigned short* wrow = w3p + (size_t)(co_base + lm) * 160;
#pragma unroll
  for (int s = 0; s < 5; ++s) A[s] = load_a_tile(wrow, s * 32 + (half << 4));

  float sAcc[8] = {0, 0, 0, 0, 0, 0, 0, 0};
  float qAcc[8] = {0, 0, 0, 0, 0, 0, 0, 0};

  for (int t = 0; t < 16; ++t) {
    const int pos = pos0 + t * 16 + lm;
    const int n   = pos / IMGHW;
    const int hw  = pos - n * IMGHW;
    const int h   = hw / HW_W;
    const int w   = hw - h * HW_W;
    const unsigned short* abase = a1 + (size_t)g * 16 * NPOS + (size_t)n * IMGHW;
    __builtin_prefetch(abase + hw + 16, 0, 3);
    fvec8 acc = {};
#pragma unroll
    for (int s = 0; s < 5; ++s) {
      bfvec16 B;
#pragma unroll
      for (int e = 0; e < 16; ++e) {
        const int kl  = (half << 4) + e;
        int tap = 2 * s + (kl >> 4);
        if (tap > 8) tap = 8;            // padded region; A is zero there
        const int ci = kl & 15;
        const int dh = tap / 3 - 1;
        const int dw = tap - (tap / 3) * 3 - 1;
        const int hh = h + dh;
        const int ww = w + dw;
        unsigned short v = 0;
        if (hh >= 0 && hh < HW_W && ww >= 0 && ww < HW_W)
          v = abase[(size_t)ci * NPOS + hh * HW_W + ww];
        B[e] = bits2bf(v);
      }
      acc = __builtin_amdgcn_wmma_f32_16x16x32_bf16(false, A[s], false, B,
                                                    (short)0, acc, false, false);
    }
    store_and_stat(acc, y3, co_base, half, pos, sAcc, qAcc);
  }
  flush_stats(sAcc, qAcc, sum3, sq3, co_base, half, lm);
}

// ---------------------------------------------------------------------------
// K5: convA 1x1 (M=256, K=512). A reloaded per K-step (weights L2-resident).
// ---------------------------------------------------------------------------
__global__ __launch_bounds__(32) void convA_kernel(
    const unsigned short* __restrict__ a3, const unsigned short* __restrict__ wab,
    unsigned short* __restrict__ y5, float* __restrict__ sumA,
    float* __restrict__ sqA) {
  const int lane = threadIdx.x & 31;
  const int half = lane >> 4;
  const int lm   = lane & 15;
  const int co_base = blockIdx.y * 16;
  const int pos0    = blockIdx.x * 256;
  const unsigned short* wrow = wab + (size_t)(co_base + lm) * 512;

  float sAcc[8] = {0, 0, 0, 0, 0, 0, 0, 0};
  float qAcc[8] = {0, 0, 0, 0, 0, 0, 0, 0};

  for (int t = 0; t < 16; ++t) {
    const int pos = pos0 + t * 16 + lm;
    __builtin_prefetch(a3 + pos + 16, 0, 3);
    fvec8 acc = {};
#pragma unroll
    for (int s = 0; s < 16; ++s) {
      bfvec16 A = load_a_tile(wrow, s * 32 + (half << 4));
      bfvec16 B = load_b_tile(a3, s * 32 + (half << 4), pos);
      acc = __builtin_amdgcn_wmma_f32_16x16x32_bf16(false, A, false, B,
                                                    (short)0, acc, false, false);
    }
    store_and_stat(acc, y5, co_base, half, pos, sAcc, qAcc);
  }
  flush_stats(sAcc, qAcc, sumA, sqA, co_base, half, lm);
}

// ---------------------------------------------------------------------------
// K7: final BN + residual + ReLU, emitted in NCHW order (fp32).
//     grid = (13, 256, 32); x covers hw in chunks of 256 (3136 guarded).
// ---------------------------------------------------------------------------
__global__ __launch_bounds__(256) void final_kernel(
    const unsigned short* __restrict__ y5, const float* __restrict__ x,
    const float* __restrict__ scaleA, const float* __restrict__ shiftA,
    float* __restrict__ out) {
  const int n  = blockIdx.z;
  const int c  = blockIdx.y;
  const int hw = blockIdx.x * 256 + threadIdx.x;
  if (hw >= IMGHW) return;
  const size_t o = ((size_t)n * 256 + c) * IMGHW + hw;
  const size_t p = (size_t)c * NPOS + (size_t)n * IMGHW + hw;
  const float v = scaleA[c] * bf2f(y5[p]) + shiftA[c] + x[o];
  out[o] = fmaxf(v, 0.0f);
}

// ---------------------------------------------------------------------------
// Workspace layout (bytes, 256-aligned):
//   [0)          w1b  512x256 bf16         262144
//   [262144)     w3p  512x160 bf16         163840
//   [425984)     wab  256x512 bf16         262144
//   [688128)     stats (2560 f32)           10240
//   [698368)     params (2560 f32)          10240
//   [708608)     y1   512xNPOS bf16     102760448
//   [103469056)  y3   512xNPOS bf16     102760448
//   [206229504)  y5   256xNPOS bf16      51380224
//   [257609728)  xb   256xNPOS bf16      51380224   (total ~309 MB)
// ---------------------------------------------------------------------------
extern "C" void kernel_launch(void* const* d_in, const int* in_sizes, int n_in,
                              void* d_out, int out_size, void* d_ws, size_t ws_size,
                              hipStream_t stream) {
  const float* x  = (const float*)d_in[0];
  const float* w1 = (const float*)d_in[1];
  const float* g1 = (const float*)d_in[2];
  const float* b1 = (const float*)d_in[3];
  const float* w3 = (const float*)d_in[4];
  const float* g3 = (const float*)d_in[5];
  const float* b3 = (const float*)d_in[6];
  const float* wa = (const float*)d_in[7];
  const float* ga = (const float*)d_in[8];
  const float* ba = (const float*)d_in[9];
  float* out = (float*)d_out;
  char* ws = (char*)d_ws;

  unsigned short* w1b = (unsigned short*)(ws + 0);
  unsigned short* w3p = (unsigned short*)(ws + 262144);
  unsigned short* wab = (unsigned short*)(ws + 425984);
  float* stats        = (float*)(ws + 688128);
  float* prm          = (float*)(ws + 698368);
  unsigned short* y1  = (unsigned short*)(ws + 708608);
  unsigned short* y3  = (unsigned short*)(ws + 103469056);
  unsigned short* y5  = (unsigned short*)(ws + 206229504);
  unsigned short* xb  = (unsigned short*)(ws + 257609728);

  float* sum1 = stats;        float* sq1 = stats + 512;
  float* sum3 = stats + 1024; float* sq3 = stats + 1536;
  float* sumA = stats + 2048; float* sqA = stats + 2304;
  float* sc1 = prm;        float* sh1 = prm + 512;
  float* sc3 = prm + 1024; float* sh3 = prm + 1536;
  float* scA = prm + 2048; float* shA = prm + 2304;

  prep_kernel<<<512, 256, 0, stream>>>(w1, w3, wa, w1b, w3p, wab, stats);
  xcvt_kernel<<<dim3(392, 256), 256, 0, stream>>>(x, xb);

  conv1_kernel<<<dim3(392, 32), 32, 0, stream>>>(xb, w1b, y1, sum1, sq1);
  bnparam_kernel<<<2, 256, 0, stream>>>(sum1, sq1, g1, b1, sc1, sh1, 512);
  bnapply_kernel<<<dim3(392, 512), 256, 0, stream>>>(y1, sc1, sh1);

  conv3_kernel<<<dim3(392, 32), 32, 0, stream>>>(y1, w3p, y3, sum3, sq3);
  bnparam_kernel<<<2, 256, 0, stream>>>(sum3, sq3, g3, b3, sc3, sh3, 512);
  bnapply_kernel<<<dim3(392, 512), 256, 0, stream>>>(y3, sc3, sh3);

  convA_kernel<<<dim3(392, 16), 32, 0, stream>>>(y3, wab, y5, sumA, sqA);
  bnparam_kernel<<<1, 256, 0, stream>>>(sumA, sqA, ga, ba, scA, shA, 256);

  final_kernel<<<dim3(13, 256, 32), 256, 0, stream>>>(y5, x, scA, shA, out);
}
